// Share_lstm_cell_51711406243888
// MI455X (gfx1250) — matrix operand order repaired and checked by reading, not compiled, not run on hardware
//
#include <hip/hip_runtime.h>
#include <cstdint>

// ---- CDNA5 WMMA types ----
typedef __attribute__((ext_vector_type(16))) __bf16 v16bf;
typedef __attribute__((ext_vector_type(8)))  float  v8f;
typedef __attribute__((ext_vector_type(4)))  int    v4i;

#define D_HID   128
#define NGATE   7
#define BATCH   65536
#define BD      (BATCH * D_HID)          // 8388608 elements per output array
#define NFRAG_EL (8 * 7 * 8 * 512)       // ntile*gate*ktile*512 bf16 = 229376
#define FRAG_BYTES_PER_NT (7 * 8 * 1024) // 57344 B = 56 KB per N-tile
#define LDS_B_U4   (FRAG_BYTES_PER_NT / 16)        // 3584 uint4 for B staging
#define LDS_TOTAL_BYTES (FRAG_BYTES_PER_NT + 8 * 12 * 1024)   // 155648 B

// CDNA5 async global->LDS staging (builtin confirmed present; exact param AS casts)
#if defined(__has_builtin)
#if __has_builtin(__builtin_amdgcn_global_load_async_to_lds_b128)
#define HAVE_ASYNC_LDS 1
#endif
#endif

#define AS_GLOBAL_I4(p) ((__attribute__((address_space(1))) v4i*)(p))
#define AS_LDS_I4(p)    ((__attribute__((address_space(3))) v4i*)(p))

// fragment union: 16 bf16 = 8 dwords = 2 uint4
union Frag {
    v16bf    bf;
    uint32_t u[8];
    uint4    q[2];
};

__device__ __forceinline__ unsigned short f2bf(float f) {
    uint32_t u = __float_as_uint(f);
    uint32_t r = u + 0x7FFFu + ((u >> 16) & 1u);   // round-to-nearest-even
    return (unsigned short)(r >> 16);
}
__device__ __forceinline__ uint32_t pk2(float lo, float hi) {
    return (uint32_t)f2bf(lo) | ((uint32_t)f2bf(hi) << 16);
}
__device__ __forceinline__ float sigm(float x) {
    return 1.0f / (1.0f + __expf(-x));
}
__device__ __forceinline__ float tanh_fast(float x) {
    float xc = fminf(fmaxf(x, -15.0f), 15.0f);
    float e  = __expf(2.0f * xc);
    return (e - 1.0f) / (e + 1.0f);
}

// Build one 16x32 bf16 A-fragment for this lane.
// ISA A-layout (16-bit, 16x32): lane L holds row M=L%16.
// lanes 0-15:  halves 0..7 = K kb..kb+7 (kb=0),  halves 8..15 = K kb+16..kb+23
// lanes 16-31: same with kb=8.
__device__ __forceinline__ void loadAfrag(Frag& fr, const float* rowptr, int off) {
    const float4* p0 = (const float4*)(rowptr + off);
    float4 a0 = p0[0];
    float4 a1 = p0[1];
    const float4* p1 = (const float4*)(rowptr + off + 16);
    float4 a2 = p1[0];
    float4 a3 = p1[1];
    fr.u[0] = pk2(a0.x, a0.y); fr.u[1] = pk2(a0.z, a0.w);
    fr.u[2] = pk2(a1.x, a1.y); fr.u[3] = pk2(a1.z, a1.w);
    fr.u[4] = pk2(a2.x, a2.y); fr.u[5] = pk2(a2.z, a2.w);
    fr.u[6] = pk2(a3.x, a3.y); fr.u[7] = pk2(a3.z, a3.w);
}

// ---------------------------------------------------------------------------
// Prologue: swizzle weights into bf16 B-fragments + fold biases.
// Frag element index: idx = ((nt*7 + g)*8 + kt)*512 + lane*16 + half
// B-layout (16-bit, 32x16): lane L -> N = L%16, K = (L<16?0:8)+half (half<8), +16+(half-8) (half>=8) within k-tile.
// pre[b,n] = sum_k A[b,k] * W[n,k]  =>  Bfrag(k,n) = W[n,k]
// ---------------------------------------------------------------------------
__global__ void prep_kernel(const float* __restrict__ wih, const float* __restrict__ bih,
                            const float* __restrict__ whh, const float* __restrict__ bhh,
                            const float* __restrict__ wsh0, const float* __restrict__ bsh0,
                            const float* __restrict__ wsh1, const float* __restrict__ bsh1,
                            const float* __restrict__ wsh2, const float* __restrict__ bsh2,
                            unsigned short* __restrict__ fragbuf,
                            float* __restrict__ biasbuf) {
    int idx = blockIdx.x * 256 + threadIdx.x;
    if (idx < NFRAG_EL) {
        int half = idx & 15;
        int lane = (idx >> 4) & 31;
        int kt   = (idx >> 9) & 7;
        int gnt  = idx >> 12;            // nt*7 + g
        int g    = gnt % 7;
        int nt   = gnt / 7;
        int n    = nt * 16 + (lane & 15);
        int k    = kt * 32 + ((lane >> 4) << 3) + (half < 8 ? half : half + 8);   // 0..255
        float val;
        if (k < 128) {
            val = wih[(g * 128 + n) * 128 + k];
        } else {
            int kk = k - 128;
            if (g < 4) {
                val = whh[(g * 128 + n) * 128 + kk];
            } else {
                const float* w = (g == 4) ? wsh0 : ((g == 5) ? wsh1 : wsh2);
                val = w[n * 128 + kk];
            }
        }
        fragbuf[idx] = f2bf(val);
    } else if (idx < NFRAG_EL + NGATE * D_HID) {
        int j = idx - NFRAG_EL;          // g*128 + n
        int g = j >> 7;
        int n = j & 127;
        float b = bih[j];
        if (g < 4)       b += bhh[j];
        else if (g == 4) b += bsh0[n];
        else if (g == 5) b += bsh1[n];
        else             b += bsh2[n];
        biasbuf[j] = b;
    }
}

// ---------------------------------------------------------------------------
// Main fused kernel: 256 threads = 8 waves; each wave owns 16 batch rows.
// LDS: [0, 56KB)           B fragments for current N-tile (shared, barriered)
//      [56KB, 56KB+96KB)   per-wave parked s0/s1/s2 A-fragments (no barrier)
// ---------------------------------------------------------------------------
__global__ __launch_bounds__(256, 1)
void lstm_main_kernel(const float* __restrict__ x,    const float* __restrict__ h,
                      const float* __restrict__ cmin, const float* __restrict__ s0,
                      const float* __restrict__ s1,   const float* __restrict__ s2,
                      const float* __restrict__ mask,
                      const unsigned short* __restrict__ fragbuf,
                      const float* __restrict__ biasbuf,
                      float* __restrict__ out) {
    extern __shared__ uint4 smem4[];

    const int tid     = threadIdx.x;
    const int lane    = tid & 31;
    const int wv      = tid >> 5;
    const int rowBase = blockIdx.x * 128 + wv * 16;
    const int r       = lane & 15;                   // A row within wave tile
    const int kb      = (lane >> 4) * 8;             // lane-half K base for A frags
    const int rowlane = rowBase + r;
    const int mrow_off = (lane >> 4) << 3;           // C-layout M offset for this lane-half

    // ---- Resident A fragments: x(4), h(4). s0/s1/s2 parked in LDS. ----
    Frag xf[4], hf[4];
    {
        const float* xr = x + rowlane * D_HID;
        const float* hr = h + rowlane * D_HID;
        const float* sr[3] = {s0 + rowlane * D_HID, s1 + rowlane * D_HID,
                              s2 + rowlane * D_HID};
#pragma unroll
        for (int t = 0; t < 4; ++t) {
            int off = t * 32 + kb;
            loadAfrag(xf[t], xr, off);
            loadAfrag(hf[t], hr, off);
#pragma unroll
            for (int j = 0; j < 3; ++j) {
                Frag tf;
                loadAfrag(tf, sr[j], off);
                int si = LDS_B_U4 + wv * 768 + (j * 4 + t) * 64 + lane * 2;
                smem4[si]     = tf.q[0];
                smem4[si + 1] = tf.q[1];
            }
        }
    }

    // ---- mask is N-invariant: hoist the 8 per-lane row masks ----
    float mval[8];
#pragma unroll
    for (int v = 0; v < 8; ++v) mval[v] = mask[rowBase + v + mrow_off];

    for (int nt = 0; nt < 8; ++nt) {
        __syncthreads();   // previous tile fully consumed (also covers s-frag parks)
        // ---- stage this N-tile's 56KB of B fragments into LDS ----
        {
            const uint4* src = (const uint4*)fragbuf + nt * LDS_B_U4;
#if HAVE_ASYNC_LDS
#pragma unroll
            for (int j = 0; j < 14; ++j) {
                int i = j * 256 + tid;
                __builtin_amdgcn_global_load_async_to_lds_b128(
                    AS_GLOBAL_I4(src + i), AS_LDS_I4(&smem4[i]), 0, 0);
            }
#if __has_builtin(__builtin_amdgcn_s_wait_asynccnt)
            __builtin_amdgcn_s_wait_asynccnt(0);
#else
            asm volatile("s_wait_asynccnt 0" ::: "memory");
#endif
#else
#pragma unroll
            for (int j = 0; j < 14; ++j) {
                int i = j * 256 + tid;
                smem4[i] = src[i];
            }
#endif
        }
        __syncthreads();

        const int col = nt * 16 + (lane & 15);

        // ---- K loop: 8 steps of 32; first step uses inline C=0 ----
        v8f acc[NGATE];
#pragma unroll
        for (int t = 0; t < 8; ++t) {
#pragma unroll
            for (int g = 0; g < NGATE; ++g) {
                Frag bfr;
                int fo = (g * 8 + t) * 64 + lane * 2;   // uint4 index into B region
                bfr.q[0] = smem4[fo];
                bfr.q[1] = smem4[fo + 1];

                Frag sfr;
                const Frag* ap;
                if (t < 4) {
                    ap = &xf[t];
                } else if (g < 4) {
                    ap = &hf[t - 4];
                } else {
                    int si = LDS_B_U4 + wv * 768 + ((g - 4) * 4 + (t - 4)) * 64 + lane * 2;
                    sfr.q[0] = smem4[si];
                    sfr.q[1] = smem4[si + 1];
                    ap = &sfr;
                }
                v8f cin = (t == 0) ? (v8f){0.f, 0.f, 0.f, 0.f, 0.f, 0.f, 0.f, 0.f}
                                   : acc[g];
                acc[g] = __builtin_amdgcn_wmma_f32_16x16x32_bf16(
                    false, ap->bf, false, bfr.bf, (short)0, cin, false, false);
            }
        }

        // ---- fused elementwise epilogue on this 16-column slice ----
        float bb[NGATE];
#pragma unroll
        for (int g = 0; g < NGATE; ++g) bb[g] = biasbuf[g * 128 + col];

#pragma unroll
        for (int v = 0; v < 8; ++v) {
            int row = rowBase + v + mrow_off;
            int off = row * D_HID + col;
            float cm = cmin[off];
            float m  = mval[v];

            float ig = sigm(acc[0][v] + bb[0]);
            float fg = sigm(acc[1][v] + bb[1]);
            float og = sigm(acc[2][v] + bb[2]);
            float ch = acc[3][v] + bb[3];

            float ct = (fg * cm + ig * tanh_fast(ch)) * m;
            float tc = tanh_fast(ct) * m;
            float hh = og * tc;

            float b0 = sigm(acc[4][v] + bb[4]);
            float b1 = sigm(acc[5][v] + bb[5]);
            float b2 = sigm(acc[6][v] + bb[6]);

            float so0 = (b0 + 1.0f - og) * tc;
            float so1 = (b1 + 2.0f - og - so0) * 0.5f * tc;
            float so2 = (b2 + 3.0f - og - (2.0f * so0 + so1)) * (1.0f / 3.0f) * tc;

            out[off]            = hh;
            out[BD + off]       = ct;
            out[2 * BD + off]   = so0;
            out[3 * BD + off]   = so1;
            out[4 * BD + off]   = so2;
        }
    }
}

extern "C" void kernel_launch(void* const* d_in, const int* in_sizes, int n_in,
                              void* d_out, int out_size, void* d_ws, size_t ws_size,
                              hipStream_t stream) {
    const float* x    = (const float*)d_in[0];
    const float* h    = (const float*)d_in[1];
    const float* cmin = (const float*)d_in[2];
    const float* s0   = (const float*)d_in[3];
    const float* s1   = (const float*)d_in[4];
    const float* s2   = (const float*)d_in[5];
    const float* mask = (const float*)d_in[6];
    const float* wih  = (const float*)d_in[7];
    const float* bih  = (const float*)d_in[8];
    const float* whh  = (const float*)d_in[9];
    const float* bhh  = (const float*)d_in[10];
    const float* wsh0 = (const float*)d_in[11];
    const float* bsh0 = (const float*)d_in[12];
    const float* wsh1 = (const float*)d_in[13];
    const float* bsh1 = (const float*)d_in[14];
    const float* wsh2 = (const float*)d_in[15];
    const float* bsh2 = (const float*)d_in[16];
    // d_in[17] = index (unused by reference math)
    float* out = (float*)d_out;

    // Workspace layout: [0,3584) bias f32 (7*128), [4096, 4096+458752) bf16 frags
    float*          biasbuf = (float*)d_ws;
    unsigned short* fragbuf = (unsigned short*)((char*)d_ws + 4096);

    {   // prologue: 229376 frag elems + 896 bias elems
        int total  = NFRAG_EL + NGATE * D_HID;
        int blocks = (total + 255) / 256;
        prep_kernel<<<blocks, 256, 0, stream>>>(wih, bih, whh, bhh, wsh0, bsh0,
                                                wsh1, bsh1, wsh2, bsh2,
                                                fragbuf, biasbuf);
    }
    {   // main: 65536 rows / 128 rows-per-block, 152KB dynamic LDS
        lstm_main_kernel<<<BATCH / 128, 256, LDS_TOTAL_BYTES, stream>>>(
            x, h, cmin, s0, s1, s2, mask, fragbuf, biasbuf, out);
    }
}
